// KanAdjunction_21036749816136
// MI455X (gfx1250) — matrix operand order repaired and checked by reading, not compiled, not run on hardware
//
#include <hip/hip_runtime.h>
#include <hip/hip_bf16.h>

// Problem sizes (fixed by the reference)
#define NSRC 8192   // N: rows of source / keys
#define NTGT 4096   // B: rows of target / queries
#define DIMF 256    // C = E = D = 256
#define SCALE 0.0625f  // 256^-0.5

typedef __bf16 bf16;
typedef __attribute__((ext_vector_type(16))) __bf16 v16bf;
typedef __attribute__((ext_vector_type(8)))  float  v8f;
typedef __attribute__((ext_vector_type(4)))  unsigned int u32x4;
typedef __attribute__((ext_vector_type(8)))  int i32x8;
typedef __attribute__((ext_vector_type(4)))  int i32x4;

union ABf { v16bf v; uint4 q[2]; };

// ---- CDNA5 feature detection (device pass only; host pass sees 0) ----
#if defined(__has_builtin)
#if __has_builtin(__builtin_amdgcn_tensor_load_to_lds)
#define HAVE_TDM 1
#endif
#endif
#if __has_include(<hip/amd_detail/amd_gfx1250_TDM.h>)
#define TDM_6ARG 1   // amdgpu-toolchain (clang-23) 6-arg builtin
#endif

__device__ __forceinline__ v8f wmma_bf16(v16bf a, v16bf b, v8f c) {
    // D = A(16x32 bf16) * B(32x16 bf16) + C(16x16 f32)
    return __builtin_amdgcn_wmma_f32_16x16x32_bf16(
        false, a, false, b, (short)0, c, false, false);
}

__device__ __forceinline__ v8f vzero() {
    v8f z;
#pragma unroll
    for (int i = 0; i < 8; ++i) z[i] = 0.0f;
    return z;
}

// max/sum across the 16 lanes of each wave32 half (xor masks stay inside the half)
__device__ __forceinline__ float half_max(float v) {
    v = fmaxf(v, __shfl_xor(v, 1));
    v = fmaxf(v, __shfl_xor(v, 2));
    v = fmaxf(v, __shfl_xor(v, 4));
    v = fmaxf(v, __shfl_xor(v, 8));
    return v;
}
__device__ __forceinline__ float half_sum(float v) {
    v += __shfl_xor(v, 1);
    v += __shfl_xor(v, 2);
    v += __shfl_xor(v, 4);
    v += __shfl_xor(v, 8);
    return v;
}

__device__ __forceinline__ void wait_tensorcnt0() {
#if defined(__has_builtin) && __has_builtin(__builtin_amdgcn_s_wait_tensorcnt)
    __builtin_amdgcn_s_wait_tensorcnt(0);
#else
    asm volatile("s_wait_tensorcnt 0x0" ::: "memory");
#endif
}

// ---------------------------------------------------------------- converts
__global__ void cvt_bf16_kernel(const float* __restrict__ in,
                                bf16* __restrict__ out, int n) {
    int i = blockIdx.x * blockDim.x + threadIdx.x;
    if (i < n) out[i] = (bf16)in[i];
}

// ---------------------------------------------------------------- row norms
__global__ void rownorm_kernel(const bf16* __restrict__ X,
                               float* __restrict__ out, int rows) {
    int gw   = (blockIdx.x * blockDim.x + threadIdx.x) >> 5;
    int lane = threadIdx.x & 31;
    if (gw >= rows) return;
    uint4 raw = *(const uint4*)(X + gw * DIMF + lane * 8);
    const bf16* e = (const bf16*)&raw;
    float s = 0.0f;
#pragma unroll
    for (int j = 0; j < 8; ++j) { float v = (float)e[j]; s += v * v; }
    s += __shfl_xor(s, 1);  s += __shfl_xor(s, 2);
    s += __shfl_xor(s, 4);  s += __shfl_xor(s, 8);  s += __shfl_xor(s, 16);
    if (lane == 0) out[gw] = s;
}

// ---------------------------------------------------------------- GEMM: out[m,e] = sum_c A[m,c]*W[e,c]
// One wave computes one 16x16 tile; K = 256 in 8 bf16 WMMAs.
template <bool BF16OUT>
__global__ __launch_bounds__(128) void gemm_nt_kernel(
    const bf16* __restrict__ A, const bf16* __restrict__ W,
    void* __restrict__ outv) {
    const int tid  = threadIdx.x;
    const int lane = tid & 31;
    const int w    = tid >> 5;
    const int n15  = lane & 15;
    const int half = lane >> 4;
    const int mbase = blockIdx.x * 16;
    const int ebase = (blockIdx.y * 4 + w) * 16;

    const bf16* arow = A + (size_t)(mbase + n15) * DIMF;
    const bf16* brow = W + (size_t)(ebase + n15) * DIMF;
    v8f acc = vzero();
#pragma unroll
    for (int cs = 0; cs < 8; ++cs) {
        ABf a, b;
        const bf16* ap = arow + cs * 32 + half * 8;   // A layout: two 8-elem runs
        a.q[0] = *(const uint4*)ap;
        a.q[1] = *(const uint4*)(ap + 16);
        const bf16* bp = brow + cs * 32 + half * 8;   // B layout: two 8-elem runs
        b.q[0] = *(const uint4*)bp;
        b.q[1] = *(const uint4*)(bp + 16);
        acc = wmma_bf16(a.v, b.v, acc);
    }
    if (BF16OUT) {
        bf16* out = (bf16*)outv;
#pragma unroll
        for (int r = 0; r < 8; ++r)
            out[(size_t)(mbase + r + 8 * half) * DIMF + ebase + n15] = (bf16)acc[r];
    } else {
        float* out = (float*)outv;
#pragma unroll
        for (int r = 0; r < 8; ++r)
            out[(size_t)(mbase + r + 8 * half) * DIMF + ebase + n15] = acc[r];
    }
}

// ---------------------------------------------------------------- fused attention
// Block = 128 threads (4 waves); each wave owns 16 query rows, full 256-dim output.
// Streams keys in tiles of 64 with online softmax. K tile staged by the Tensor
// Data Mover (TDM) when available; F tile transposed manually in parallel.
template <bool RIGHT>
__global__ __launch_bounds__(128) void attn_kernel(
    const bf16* __restrict__ Q,  const bf16* __restrict__ Kc,
    const bf16* __restrict__ Fc, const float* __restrict__ t2,
    const float* __restrict__ kn2, float* __restrict__ out) {
    __shared__ bf16 Kt[64][272];     // key tile row-major; 544B rows (= 512B data + 32B TDM pad)
    __shared__ bf16 Ft[256][72];     // value tile transposed: Ft[d][k], 144B rows
    __shared__ bf16 Pw[4][16][72];   // per-wave P staging for C/D -> A relayout

    const int tid  = threadIdx.x;
    const int lane = tid & 31;
    const int w    = tid >> 5;
    const int n15  = lane & 15;
    const int half = lane >> 4;
    const int qbase = blockIdx.x * 64 + w * 16;
    const int qr    = qbase + n15;

    // Q held in registers as 8 A-operands (E = 256 = 8 * 32)
    ABf qa[8];
#pragma unroll
    for (int es = 0; es < 8; ++es) {
        const bf16* qp = Q + (size_t)qr * DIMF + es * 32 + half * 8;
        qa[es].q[0] = *(const uint4*)qp;
        qa[es].q[1] = *(const uint4*)(qp + 16);
    }
    float t2v[8];
#pragma unroll
    for (int r = 0; r < 8; ++r) t2v[r] = 0.0f;
    if (RIGHT) {
#pragma unroll
        for (int r = 0; r < 8; ++r) t2v[r] = t2[qbase + r + 8 * half];
    }

    v8f o[16];
#pragma unroll
    for (int t = 0; t < 16; ++t) o[t] = vzero();
    float mrow[8], lrow[8];
#pragma unroll
    for (int r = 0; r < 8; ++r) { mrow[r] = -3.0e38f; lrow[r] = 0.0f; }

#if HAVE_TDM
    // Constant part of the K-tile TDM descriptor (all values wave-uniform).
    // 2D tensor, 4-byte units: rows of 128 dwords, tile 128x64, pad 8 dwords
    // after every 128 dwords (-> 544B LDS row stride).
    const unsigned int kt_lds_base = (unsigned int)(uintptr_t)&Kt[0][0];
    i32x8 g1;
    g1[0] = (int)((2u << 16) | (1u << 20) | (6u << 22) | (7u << 25)); // data_size=4B, pad_en, interval=128dw, amount=8dw
    g1[1] = (int)((128u & 0xFFFFu) << 16);          // tensor_dim0 = 128 (low 16)
    g1[2] = (int)((128u >> 16) | ((unsigned)NSRC << 16)); // dim0 hi | tensor_dim1 lo
    g1[3] = (int)(((unsigned)NSRC >> 16) | (128u << 16)); // dim1 hi | tile_dim0 = 128
    g1[4] = (int)(64u);                              // tile_dim1 = 64 rows, tile_dim2 = 0
    g1[5] = (int)(128u);                             // tensor_dim0_stride = 128 dwords
    g1[6] = 0;
    g1[7] = 0;
    i32x4 zz; zz[0] = 0; zz[1] = 0; zz[2] = 0; zz[3] = 0;
#endif

    for (int kb = 0; kb < NSRC; kb += 64) {
        __syncthreads();  // previous iteration done with Kt/Ft

        // ---- K tile: TDM DMA (wave 0) overlapped with F transpose staging
#if HAVE_TDM
        if (w == 0) {
            unsigned long long ga = (unsigned long long)(uintptr_t)(Kc + (size_t)kb * DIMF);
            u32x4 g0;
            g0[0] = 1u;                              // count=1, user mode
            g0[1] = kt_lds_base;                     // LDS byte offset
            g0[2] = (unsigned int)ga;
            g0[3] = (unsigned int)((ga >> 32) & 0x1FFFFFFu) | (2u << 30); // type=2 (image)
#if TDM_6ARG
            __builtin_amdgcn_tensor_load_to_lds(g0, g1, zz, zz, g1, 0);
#else
            __builtin_amdgcn_tensor_load_to_lds(g0, g1, zz, zz, 0);
#endif
        }
#else
#pragma unroll
        for (int i = 0; i < 16; ++i) {
            int u = i * 128 + tid;              // 0..2047 -> 64 rows x 32 chunks
            int row = u >> 5, c8 = (u & 31) * 8;
            *(uint4*)&Kt[row][c8] = *(const uint4*)(Kc + (size_t)(kb + row) * DIMF + c8);
        }
#endif

        // ---- F tile transposed: two consecutive key rows packed per b32 store
#pragma unroll
        for (int i = 0; i < 8; ++i) {
            int u  = i * 128 + tid;             // 0..1023
            int kp = u >> 5;                    // key-row pair 0..31
            int d8 = (u & 31) * 8;
            uint4 r0 = *(const uint4*)(Fc + (size_t)(kb + 2 * kp)     * DIMF + d8);
            uint4 r1 = *(const uint4*)(Fc + (size_t)(kb + 2 * kp + 1) * DIMF + d8);
            const unsigned short* e0 = (const unsigned short*)&r0;
            const unsigned short* e1 = (const unsigned short*)&r1;
#pragma unroll
            for (int j = 0; j < 8; ++j)
                *(unsigned int*)&Ft[d8 + j][2 * kp] =
                    (unsigned int)e0[j] | ((unsigned int)e1[j] << 16);
        }

        // prefetch next tiles into cache while we compute
        if (kb + 64 < NSRC) {
            __builtin_prefetch(Kc + (size_t)(kb + 64 + (tid >> 1)) * DIMF + (tid & 1) * 128, 0, 0);
            __builtin_prefetch(Fc + (size_t)(kb + 64 + (tid >> 1)) * DIMF + (tid & 1) * 128, 0, 0);
        }

#if HAVE_TDM
        if (w == 0) wait_tensorcnt0();
#endif
        __syncthreads();

        // ---- S = Q K^T  (16 x 64), f32 accum: 32 WMMAs
        v8f s[4];
#pragma unroll
        for (int nt = 0; nt < 4; ++nt) s[nt] = vzero();
#pragma unroll
        for (int es = 0; es < 8; ++es) {
#pragma unroll
            for (int nt = 0; nt < 4; ++nt) {
                ABf b;
                const bf16* bp = &Kt[nt * 16 + n15][es * 32 + half * 8];
                b.q[0] = *(const uint4*)bp;
                b.q[1] = *(const uint4*)(bp + 16);
                s[nt] = wmma_bf16(qa[es].v, b.v, s[nt]);
            }
        }

        // ---- logits
        float lg[4][8];
        if (RIGHT) {
#pragma unroll
            for (int nt = 0; nt < 4; ++nt) {
                float kn = kn2[kb + nt * 16 + n15];
#pragma unroll
                for (int r = 0; r < 8; ++r) {
                    float d2 = t2v[r] + kn - 2.0f * s[nt][r];
                    lg[nt][r] = -SCALE * sqrtf(fmaxf(d2, 0.0f));
                }
            }
        } else {
#pragma unroll
            for (int nt = 0; nt < 4; ++nt)
#pragma unroll
                for (int r = 0; r < 8; ++r) lg[nt][r] = SCALE * s[nt][r];
        }

        // ---- online softmax (row stats live in the C/D accumulator lane layout)
        float alpha[8];
#pragma unroll
        for (int r = 0; r < 8; ++r) {
            float tm = fmaxf(fmaxf(lg[0][r], lg[1][r]), fmaxf(lg[2][r], lg[3][r]));
            tm = half_max(tm);
            float mn = fmaxf(mrow[r], tm);
            alpha[r] = __expf(mrow[r] - mn);
            mrow[r]  = mn;
            float ps = 0.0f;
#pragma unroll
            for (int nt = 0; nt < 4; ++nt) {
                float p = __expf(lg[nt][r] - mn);
                ps += p;
                Pw[w][r + 8 * half][nt * 16 + n15] = (bf16)p;
            }
            lrow[r] = lrow[r] * alpha[r] + half_sum(ps);
        }
#pragma unroll
        for (int t = 0; t < 16; ++t)
#pragma unroll
            for (int r = 0; r < 8; ++r) o[t][r] *= alpha[r];

        // wave-private LDS round trip: DS wait is enough (no workgroup barrier)
        asm volatile("s_wait_dscnt 0x0" ::: "memory");

        // ---- O += P (16x64) * F (64x256): 32 WMMAs
        ABf pa[2];
#pragma unroll
        for (int c = 0; c < 2; ++c) {
            const bf16* pp = &Pw[w][n15][c * 32 + half * 8];
            pa[c].q[0] = *(const uint4*)pp;
            pa[c].q[1] = *(const uint4*)(pp + 16);
        }
#pragma unroll
        for (int t = 0; t < 16; ++t) {
#pragma unroll
            for (int c = 0; c < 2; ++c) {
                ABf b;
                const bf16* fp = &Ft[t * 16 + n15][c * 32 + half * 8];
                b.q[0] = *(const uint4*)fp;
                b.q[1] = *(const uint4*)(fp + 16);
                o[t] = wmma_bf16(pa[c].v, b.v, o[t]);
            }
        }
    }

    // ---- epilogue: normalize and store
    float rinv[8];
#pragma unroll
    for (int r = 0; r < 8; ++r) rinv[r] = 1.0f / lrow[r];
#pragma unroll
    for (int t = 0; t < 16; ++t)
#pragma unroll
        for (int r = 0; r < 8; ++r)
            out[(size_t)(qbase + r + 8 * half) * DIMF + t * 16 + n15] = o[t][r] * rinv[r];
}

// ---------------------------------------------------------------- launcher
extern "C" void kernel_launch(void* const* d_in, const int* in_sizes, int n_in,
                              void* d_out, int out_size, void* d_ws, size_t ws_size,
                              hipStream_t stream) {
    const float* source = (const float*)d_in[0];  // [8192,256]
    const float* target = (const float*)d_in[1];  // [4096,256]
    const float* WKl    = (const float*)d_in[2];  // [256,256]
    const float* WFl    = (const float*)d_in[3];
    const float* WKr    = (const float*)d_in[4];
    const float* WFr    = (const float*)d_in[5];
    const float* Wr     = (const float*)d_in[6];
    float* out = (float*)d_out;  // [left | right | restriction], each 4096*256 f32

    // workspace layout (bf16 copies + projections + norms) ~22.7 MB
    char* p = (char*)d_ws;
    bf16* src_bf = (bf16*)p;  p += (size_t)NSRC * DIMF * 2;
    bf16* tgt_bf = (bf16*)p;  p += (size_t)NTGT * DIMF * 2;
    bf16* WKl_bf = (bf16*)p;  p += (size_t)DIMF * DIMF * 2;
    bf16* WFl_bf = (bf16*)p;  p += (size_t)DIMF * DIMF * 2;
    bf16* WKr_bf = (bf16*)p;  p += (size_t)DIMF * DIMF * 2;
    bf16* WFr_bf = (bf16*)p;  p += (size_t)DIMF * DIMF * 2;
    bf16* Wr_bf  = (bf16*)p;  p += (size_t)DIMF * DIMF * 2;
    bf16* Kc_l   = (bf16*)p;  p += (size_t)NSRC * DIMF * 2;
    bf16* Fc_l   = (bf16*)p;  p += (size_t)NSRC * DIMF * 2;
    bf16* Kc_r   = (bf16*)p;  p += (size_t)NSRC * DIMF * 2;
    bf16* Fc_r   = (bf16*)p;  p += (size_t)NSRC * DIMF * 2;
    float* kn2   = (float*)p; p += (size_t)NSRC * 4;
    float* t2    = (float*)p; p += (size_t)NTGT * 4;

    // 1) f32 -> bf16 converts
    auto cvt = [&](const float* in, bf16* o, int n) {
        cvt_bf16_kernel<<<(n + 255) / 256, 256, 0, stream>>>(in, o, n);
    };
    cvt(source, src_bf, NSRC * DIMF);
    cvt(target, tgt_bf, NTGT * DIMF);
    cvt(WKl, WKl_bf, DIMF * DIMF);
    cvt(WFl, WFl_bf, DIMF * DIMF);
    cvt(WKr, WKr_bf, DIMF * DIMF);
    cvt(WFr, WFr_bf, DIMF * DIMF);
    cvt(Wr,  Wr_bf,  DIMF * DIMF);

    // 2) projections: Kc/Fc = source @ W^T  (bf16 out)
    dim3 gproj(NSRC / 16, 4);
    gemm_nt_kernel<true><<<gproj, 128, 0, stream>>>(src_bf, WKl_bf, Kc_l);
    gemm_nt_kernel<true><<<gproj, 128, 0, stream>>>(src_bf, WFl_bf, Fc_l);
    gemm_nt_kernel<true><<<gproj, 128, 0, stream>>>(src_bf, WKr_bf, Kc_r);
    gemm_nt_kernel<true><<<gproj, 128, 0, stream>>>(src_bf, WFr_bf, Fc_r);

    // 3) row norms for the cdist branch
    rownorm_kernel<<<NSRC * 32 / 256, 256, 0, stream>>>(Kc_r, kn2, NSRC);
    rownorm_kernel<<<NTGT * 32 / 256, 256, 0, stream>>>(tgt_bf, t2, NTGT);

    // 4) fused attention (left: dot-softmax; right: -scale*cdist softmax)
    attn_kernel<false><<<NTGT / 64, 128, 0, stream>>>(tgt_bf, Kc_l, Fc_l, t2, kn2, out);
    attn_kernel<true><<<NTGT / 64, 128, 0, stream>>>(tgt_bf, Kc_r, Fc_r, t2, kn2,
                                                     out + (size_t)NTGT * DIMF);

    // 5) restriction = target @ W_restrict^T (f32 straight to d_out)
    dim3 grst(NTGT / 16, 4);
    gemm_nt_kernel<false><<<grst, 128, 0, stream>>>(tgt_bf, Wr_bf,
                                                    out + 2 * (size_t)NTGT * DIMF);
}
